// HRMChess_8933531976442
// MI455X (gfx1250) — matrix operand order repaired and checked by reading, not compiled, not run on hardware
//
#include <hip/hip_runtime.h>
#include <hip/hip_bf16.h>

// ---------------------------------------------------------------------------
// Types
// ---------------------------------------------------------------------------
typedef __attribute__((ext_vector_type(16))) __bf16 bf16x16;
typedef __attribute__((ext_vector_type(8)))  __bf16 bf16x8;
typedef __attribute__((ext_vector_type(8)))  float  f32x8;

__device__ __forceinline__ __bf16 to_bf16(float f) {
    unsigned u = __float_as_uint(f);
    u += 0x7fffu + ((u >> 16) & 1u);           // round-to-nearest-even
    unsigned short s = (unsigned short)(u >> 16);
    __bf16 r;
    __builtin_memcpy(&r, &s, sizeof(r));
    return r;
}

__device__ __forceinline__ f32x8 wmma_bf16(bf16x16 a, bf16x16 b, f32x8 c) {
    // D = A(16x32) * B(32x16) + C, f32 accumulate
    return __builtin_amdgcn_wmma_f32_16x16x32_bf16(false, a, false, b,
                                                   (short)0, c, false, false);
}

__device__ __forceinline__ bf16x16 load_a_frag(const __bf16* p) {
    // A-layout: per-lane 16 contiguous elements (two 16B segments at +0,+8)
    union { bf16x16 v; bf16x8 h[2]; } u;
    u.h[0] = *(const bf16x8*)(p);
    u.h[1] = *(const bf16x8*)(p + 8);
    return u.v;
}

enum {
    FLAG_RELU  = 1,
    FLAG_ACC   = 2,   // C += (read f32 C first)
    FLAG_WF32  = 4,   // write f32 C
    FLAG_WBF16 = 8,   // write bf16 copy
};

// ---------------------------------------------------------------------------
// Generic bf16 WMMA GEMM:  C[M,N] = A[M,K] * W[N,K]^T (+bias)(relu)
// block = (32,4): 4 waves in 2x2, each wave a 32x32 tile (4 wmma / K-step).
// Block covers 64Mx64N. Requires M%64==0, N%64==0, K%32==0,
// lda/ldb multiples of 16 elements.
// CONV: remap A rows by (dy,dx) board shift (3x3 conv on 8x8 boards).
// ---------------------------------------------------------------------------
template<bool CONV>
__global__ __launch_bounds__(128)
void gemm_bf16_kernel(const __bf16* __restrict__ A, long lda,
                      const __bf16* __restrict__ Bw, long ldb,
                      const float* __restrict__ bias,
                      float* __restrict__ Cf, long ldc,
                      __bf16* __restrict__ Cb, long ldcb,
                      int M, int N, int K, int flags, int dy, int dx)
{
    const int lane = threadIdx.x;
    const int wave = threadIdx.y;
    const int m0 = blockIdx.y * 64 + (wave >> 1) * 32;
    const int n0 = blockIdx.x * 64 + (wave & 1) * 32;

    const int lrow  = lane & 15;
    const int kaoff = (lane < 16) ? 0 : 16;  // A-fragment K base per lane half
    const int kboff = (lane < 16) ? 0 : 16;  // B-fragment K base per lane half

    // ---- A row pointers (2 sub-tiles in M), with conv board-shift remap ----
    long rowA0 = m0 + lrow;
    long rowA1 = m0 + 16 + lrow;
    bool valid0 = true, valid1 = true;
    if (CONV) {
        {
            int r = m0 + lrow;
            int p = r & 63;
            int y = (p >> 3) + dy, x = (p & 7) + dx;
            valid0 = ((unsigned)y < 8u) && ((unsigned)x < 8u);
            rowA0 = valid0 ? (long)((r & ~63) | (y * 8 + x)) : 0;
        }
        {
            int r = m0 + 16 + lrow;
            int p = r & 63;
            int y = (p >> 3) + dy, x = (p & 7) + dx;
            valid1 = ((unsigned)y < 8u) && ((unsigned)x < 8u);
            rowA1 = valid1 ? (long)((r & ~63) | (y * 8 + x)) : 0;
        }
    }
    const __bf16* ap0 = A + rowA0 * lda + kaoff;
    const __bf16* ap1 = A + rowA1 * lda + kaoff;
    const __bf16* bp0 = Bw + (long)(n0 + lrow) * ldb + kboff;
    const __bf16* bp1 = Bw + (long)(n0 + 16 + lrow) * ldb + kboff;

    // ---- accumulators: 2x2 tiles of 16x16 ----
    const int crow = (lane < 16) ? 0 : 8;
    f32x8 acc[2][2];
    if (flags & FLAG_ACC) {
        for (int tm = 0; tm < 2; tm++)
            for (int tn = 0; tn < 2; tn++) {
                int cr = m0 + tm * 16 + crow;
                int cc = n0 + tn * 16 + lrow;
                for (int e = 0; e < 8; e++)
                    acc[tm][tn][e] = Cf[(long)(cr + e) * ldc + cc];
            }
    } else {
        for (int tm = 0; tm < 2; tm++)
            for (int tn = 0; tn < 2; tn++)
                for (int e = 0; e < 8; e++)
                    acc[tm][tn][e] = 0.0f;
    }

    bf16x16 zfrag;
    if (CONV) {
        const __bf16 bz = to_bf16(0.0f);
        for (int e = 0; e < 16; e++) zfrag[e] = bz;
    }

    // ---- K loop: 4 wmma per step ----
    for (int k0 = 0; k0 < K; k0 += 32) {
        if (!CONV && (k0 + 256) < K) {
            // A is the streaming operand (weights stay L2-resident);
            // prefetch ahead (lowers to global_prefetch_b8)
            __builtin_prefetch((const void*)(ap0 + k0 + 256), 0, 0);
            __builtin_prefetch((const void*)(ap1 + k0 + 256), 0, 0);
        }
        bf16x16 a0, a1;
        if (CONV) {
            a0 = valid0 ? load_a_frag(ap0 + k0) : zfrag;
            a1 = valid1 ? load_a_frag(ap1 + k0) : zfrag;
        } else {
            a0 = load_a_frag(ap0 + k0);
            a1 = load_a_frag(ap1 + k0);
        }
        bf16x16 b0 = *(const bf16x16*)(bp0 + k0);
        bf16x16 b1 = *(const bf16x16*)(bp1 + k0);
        acc[0][0] = wmma_bf16(a0, b0, acc[0][0]);
        acc[0][1] = wmma_bf16(a0, b1, acc[0][1]);
        acc[1][0] = wmma_bf16(a1, b0, acc[1][0]);
        acc[1][1] = wmma_bf16(a1, b1, acc[1][1]);
    }

    // ---- epilogue ----
    for (int tn = 0; tn < 2; tn++) {
        const int cc = n0 + tn * 16 + lrow;
        const float bv = bias ? bias[cc] : 0.0f;
        for (int tm = 0; tm < 2; tm++) {
            const int cr = m0 + tm * 16 + crow;
            for (int e = 0; e < 8; e++) {
                float v = acc[tm][tn][e] + bv;
                if (flags & FLAG_RELU) v = fmaxf(v, 0.0f);
                const long rI = (long)(cr + e);
                if (flags & FLAG_WF32)  Cf[rI * ldc + cc] = v;
                if (flags & FLAG_WBF16) Cb[rI * ldcb + cc] = to_bf16(v);
            }
        }
    }
    (void)M; (void)N;
}

// ---------------------------------------------------------------------------
// conv1: 20 -> 256 channels, 3x3 SAME on 8x8 boards, relu, bf16 token-major out
// out[(b*64+p)*256 + o]
// ---------------------------------------------------------------------------
__global__ __launch_bounds__(256)
void conv1_kernel(const float* __restrict__ bp, const float* __restrict__ w,
                  const float* __restrict__ bvec, __bf16* __restrict__ out)
{
    long idx = (long)blockIdx.x * blockDim.x + threadIdx.x;
    const long total = (long)2048 * 64 * 256;
    if (idx >= total) return;
    int o = (int)(idx & 255);
    long m = idx >> 8;
    int p = (int)(m & 63);
    long b = m >> 6;
    int y = p >> 3, x = p & 7;
    float s = bvec[o];
    const float* wp = w + (long)o * 20 * 9;
    const float* ip = bp + b * 20 * 64;
    for (int c = 0; c < 20; c++) {
        for (int ddy = -1; ddy <= 1; ddy++) {
            int yy = y + ddy;
            if ((unsigned)yy >= 8u) continue;
            for (int ddx = -1; ddx <= 1; ddx++) {
                int xx = x + ddx;
                if ((unsigned)xx >= 8u) continue;
                s += ip[c * 64 + yy * 8 + xx] * wp[c * 9 + (ddy + 1) * 3 + (ddx + 1)];
            }
        }
    }
    out[idx] = to_bf16(fmaxf(s, 0.0f));
}

// ---------------------------------------------------------------------------
// f32 -> bf16 conversion (weights / generic)
// ---------------------------------------------------------------------------
__global__ void cvt_kernel(const float* __restrict__ s, __bf16* __restrict__ d, long n)
{
    long i = (long)blockIdx.x * blockDim.x + threadIdx.x;
    long stride = (long)gridDim.x * blockDim.x;
    for (; i < n; i += stride) d[i] = to_bf16(s[i]);
}

// conv2_w [O,I,3,3] f32 -> 9 contiguous bf16 matrices out[off][o*256+i]
__global__ void repack_conv2_kernel(const float* __restrict__ w, __bf16* __restrict__ out)
{
    long idx = (long)blockIdx.x * blockDim.x + threadIdx.x;
    const long total = (long)256 * 256 * 9;
    if (idx >= total) return;
    int off = (int)(idx % 9);
    long oi = idx / 9;
    out[(long)off * 65536 + oi] = to_bf16(w[idx]);
}

// ---------------------------------------------------------------------------
// Fused attention: one workgroup per (batch, head). 256 threads = 8 waves.
// qkv f32 [B,64,768]; out bf16 [B,64,256] (head-sliced write).
// ---------------------------------------------------------------------------
__global__ __launch_bounds__(256)
void attn_kernel(const float* __restrict__ qkv, __bf16* __restrict__ ob)
{
    __shared__ __bf16 Qs[64 * 64];   // Q tile (later reused as P, bf16 probs)
    __shared__ __bf16 Ks[64 * 64];
    __shared__ __bf16 Vs[64 * 64];
    __shared__ float  Sf[64 * 64];   // scores

    const int bh = blockIdx.x;       // 0 .. 8191
    const long b = bh >> 2;
    const int  h = bh & 3;
    const int tid = threadIdx.x;
    const float* base = qkv + b * 64 * 768;

    for (int i = tid; i < 4096; i += 256) {
        int t = i >> 6, d = i & 63;
        const float* rp = base + (long)t * 768 + h * 64 + d;
        Qs[i] = to_bf16(rp[0]);
        Ks[i] = to_bf16(rp[256]);
        Vs[i] = to_bf16(rp[512]);
    }
    __syncthreads();

    const int lane = tid & 31;
    const int wave = tid >> 5;
    const int kaoff = (lane < 16) ? 0 : 16;
    const int kboff = (lane < 16) ? 0 : 16;
    const int lrow = lane & 15;
    const int crow = (lane < 16) ? 0 : 8;

    // S = (Q K^T) * 1/sqrt(64)
    for (int tile = wave; tile < 16; tile += 8) {
        int ti = tile >> 2, tj = tile & 3;
        f32x8 acc; for (int e = 0; e < 8; e++) acc[e] = 0.0f;
        int arow = ti * 16 + lrow;
        int bcol = tj * 16 + lrow;
        for (int k0 = 0; k0 < 64; k0 += 32) {
            bf16x16 a, bm;
            for (int e = 0; e < 8; e++) {
                a[e]     = Qs[arow * 64 + k0 + kaoff + e];
                a[8 + e] = Qs[arow * 64 + k0 + kaoff + 8 + e];
            }
            for (int e = 0; e < 16; e++)
                bm[e] = Ks[bcol * 64 + k0 + kboff + e];
            acc = wmma_bf16(a, bm, acc);
        }
        int cmr = ti * 16 + crow;
        for (int e = 0; e < 8; e++)
            Sf[(cmr + e) * 64 + tj * 16 + lrow] = acc[e] * 0.125f;
    }
    __syncthreads();

    // softmax per row (64 rows, one thread each); write probs as bf16 into Qs
    if (tid < 64) {
        float mx = -1e30f;
        for (int j = 0; j < 64; j++) mx = fmaxf(mx, Sf[tid * 64 + j]);
        float s = 0.0f;
        for (int j = 0; j < 64; j++) {
            float e = __expf(Sf[tid * 64 + j] - mx);
            Sf[tid * 64 + j] = e;
            s += e;
        }
        float inv = 1.0f / s;
        for (int j = 0; j < 64; j++)
            Qs[tid * 64 + j] = to_bf16(Sf[tid * 64 + j] * inv);
    }
    __syncthreads();

    // O = P V   (P in Qs)
    for (int tile = wave; tile < 16; tile += 8) {
        int ti = tile >> 2, tj = tile & 3;
        f32x8 acc; for (int e = 0; e < 8; e++) acc[e] = 0.0f;
        int arow = ti * 16 + lrow;
        int bcol = tj * 16 + lrow;
        for (int k0 = 0; k0 < 64; k0 += 32) {
            bf16x16 a, bm;
            for (int e = 0; e < 8; e++) {
                a[e]     = Qs[arow * 64 + k0 + kaoff + e];
                a[8 + e] = Qs[arow * 64 + k0 + kaoff + 8 + e];
            }
            for (int e = 0; e < 16; e++)
                bm[e] = Vs[(k0 + kboff + e) * 64 + bcol];   // B non-transposed
            acc = wmma_bf16(a, bm, acc);
        }
        int cmr = ti * 16 + crow;
        for (int e = 0; e < 8; e++) {
            long tok = cmr + e;
            ob[(b * 64 + tok) * 256 + h * 64 + tj * 16 + lrow] = to_bf16(acc[e]);
        }
    }
}

// ---------------------------------------------------------------------------
// LayerNorm over 256 columns. One wave32 per row (8 cols/lane).
// out = (y [+ resid] - mean)/sqrt(var+eps) * g + b
// f32 out optional; up to two strided bf16 outputs (for concat buffers).
// ---------------------------------------------------------------------------
__global__ __launch_bounds__(256)
void ln256_kernel(const float* __restrict__ Y, const float* __restrict__ R,
                  const float* __restrict__ g, const float* __restrict__ bt,
                  float* __restrict__ outF,
                  __bf16* __restrict__ o0, long ld0,
                  __bf16* __restrict__ o1, long ld1, int M)
{
    int row = blockIdx.x * 8 + threadIdx.y;
    if (row >= M) return;
    int lane = threadIdx.x;
    long base = (long)row * 256;

    float v[8];
    for (int j = 0; j < 8; j++) {
        int c = lane + 32 * j;
        float t = Y[base + c];
        if (R) t += R[base + c];
        v[j] = t;
    }
    float s = 0.0f;
    for (int j = 0; j < 8; j++) s += v[j];
    for (int o = 16; o > 0; o >>= 1) s += __shfl_xor(s, o, 32);
    float mean = s * (1.0f / 256.0f);
    float q = 0.0f;
    for (int j = 0; j < 8; j++) { float d = v[j] - mean; q += d * d; }
    for (int o = 16; o > 0; o >>= 1) q += __shfl_xor(q, o, 32);
    float inv = rsqrtf(q * (1.0f / 256.0f) + 1e-5f);

    for (int j = 0; j < 8; j++) {
        int c = lane + 32 * j;
        float ov = (v[j] - mean) * inv * g[c] + bt[c];
        if (outF) outF[base + c] = ov;
        if (o0) o0[(long)row * ld0 + c] = to_bf16(ov);
        if (o1) o1[(long)row * ld1 + c] = to_bf16(ov);
    }
}

// ---------------------------------------------------------------------------
// Host-side orchestration
// ---------------------------------------------------------------------------
extern "C" void kernel_launch(void* const* d_in, const int* in_sizes, int n_in,
                              void* d_out, int out_size, void* d_ws, size_t ws_size,
                              hipStream_t stream)
{
    (void)in_sizes; (void)n_in; (void)out_size; (void)ws_size;

    const long BATCH = 2048;
    const long MT    = BATCH * 64;       // 131072 token rows

    // ---- inputs (setup_inputs order) ----
    const float* bitplanes = (const float*)d_in[0];
    const float* conv1_w   = (const float*)d_in[1];
    const float* conv1_b   = (const float*)d_in[2];
    const float* conv2_w   = (const float*)d_in[3];
    const float* conv2_b   = (const float*)d_in[4];
    const float* in_proj_w = (const float*)d_in[5];
    const float* in_proj_b = (const float*)d_in[6];
    const float* out_proj_w= (const float*)d_in[7];
    const float* out_proj_b= (const float*)d_in[8];
    const float* ln1_g = (const float*)d_in[9],  *ln1_b = (const float*)d_in[10];
    const float* ff1_w = (const float*)d_in[11], *ff1_b = (const float*)d_in[12];
    const float* ff2_w = (const float*)d_in[13], *ff2_b = (const float*)d_in[14];
    const float* ln2_g = (const float*)d_in[15], *ln2_b = (const float*)d_in[16];
    const float* be1_w = (const float*)d_in[17], *be1_b = (const float*)d_in[18];
    const float* be2_w = (const float*)d_in[19], *be2_b = (const float*)d_in[20];
    const float* L1_w  = (const float*)d_in[21], *L1_b  = (const float*)d_in[22];
    const float* L2_w  = (const float*)d_in[23], *L2_b  = (const float*)d_in[24];
    const float* Lln_g = (const float*)d_in[25], *Lln_b = (const float*)d_in[26];
    const float* H1_w  = (const float*)d_in[27], *H1_b  = (const float*)d_in[28];
    const float* H2_w  = (const float*)d_in[29], *H2_b  = (const float*)d_in[30];
    const float* Hln_g = (const float*)d_in[31], *Hln_b = (const float*)d_in[32];
    const float* v1_w  = (const float*)d_in[33], *v1_b  = (const float*)d_in[34];
    const float* v2_w  = (const float*)d_in[35], *v2_b  = (const float*)d_in[36];

    // ---- scratch bump allocator ----
    char* wsb = (char*)d_ws;
    size_t off = 0;
    auto alloc = [&](size_t bytes) -> void* {
        off = (off + 255) & ~(size_t)255;
        void* p = wsb + off;
        off += bytes;
        return p;
    };

    // f32 activations
    float* qkvf  = (float*)alloc(MT * 768 * sizeof(float));     // qkv
    float* x0f   = (float*)alloc(MT * 256 * sizeof(float));     // conv2 out (residual 1)
    float* tmpf  = (float*)alloc(MT * 256 * sizeof(float));     // out_proj / ff2 out
    float* x1f   = (float*)alloc(MT * 256 * sizeof(float));     // post-LN1 (residual 2)
    float* tmp2f = (float*)alloc(BATCH * 256 * sizeof(float));  // loop GEMM2 out

    // bf16 weights
    __bf16* w2p   = (__bf16*)alloc((size_t)9 * 65536 * 2);
    __bf16* inpw  = (__bf16*)alloc((size_t)768 * 256 * 2);
    __bf16* outpw = (__bf16*)alloc((size_t)256 * 256 * 2);
    __bf16* ff1w  = (__bf16*)alloc((size_t)512 * 256 * 2);
    __bf16* ff2w  = (__bf16*)alloc((size_t)256 * 512 * 2);
    __bf16* be1w  = (__bf16*)alloc((size_t)256 * 16384 * 2);
    __bf16* be2w  = (__bf16*)alloc((size_t)256 * 256 * 2);
    __bf16* L1wB  = (__bf16*)alloc((size_t)256 * 768 * 2);
    __bf16* L2wB  = (__bf16*)alloc((size_t)256 * 256 * 2);
    __bf16* H1wB  = (__bf16*)alloc((size_t)256 * 512 * 2);
    __bf16* H2wB  = (__bf16*)alloc((size_t)256 * 256 * 2);
    __bf16* v1wB  = (__bf16*)alloc((size_t)256 * 256 * 2);
    __bf16* v2wB  = (__bf16*)alloc((size_t)128 * 256 * 2);

    // bf16 activations
    __bf16* c1b   = (__bf16*)alloc((size_t)MT * 256 * 2);   // conv1 out
    __bf16* x0b   = (__bf16*)alloc((size_t)MT * 256 * 2);   // conv2 out
    __bf16* obb   = (__bf16*)alloc((size_t)MT * 256 * 2);   // attention out
    __bf16* x1b   = (__bf16*)alloc((size_t)MT * 256 * 2);   // post-LN1
    __bf16* x2b   = (__bf16*)alloc((size_t)MT * 256 * 2);   // post-LN2 (== xf [2048,16384])
    __bf16* ffh   = (__bf16*)alloc((size_t)MT * 512 * 2);   // ff hidden
    __bf16* beh   = (__bf16*)alloc((size_t)BATCH * 256 * 2);// be hidden
    __bf16* catL  = (__bf16*)alloc((size_t)BATCH * 768 * 2);// [zL | zH | be]
    __bf16* catH  = (__bf16*)alloc((size_t)BATCH * 512 * 2);// [zH | zL]
    __bf16* hidb  = (__bf16*)alloc((size_t)BATCH * 256 * 2);// loop hidden
    __bf16* hid2b = (__bf16*)alloc((size_t)BATCH * 256 * 2);// head hidden

    float* outF = (float*)d_out;  // [2048,128]

    // ---- helpers ----
    auto cvt = [&](const float* s, __bf16* d, long n) {
        cvt_kernel<<<1024, 256, 0, stream>>>(s, d, n);
    };
    auto gemm = [&](const __bf16* A, long lda, const __bf16* Bw, long ldb,
                    const float* bias, float* Cf, long ldc,
                    __bf16* Cb, long ldcb, int M, int N, int K, int flags) {
        dim3 grid((unsigned)(N / 64), (unsigned)(M / 64));
        dim3 block(32, 4);
        gemm_bf16_kernel<false><<<grid, block, 0, stream>>>(A, lda, Bw, ldb, bias,
                                                            Cf, ldc, Cb, ldcb,
                                                            M, N, K, flags, 0, 0);
    };
    auto gemm_conv = [&](const __bf16* A, const __bf16* Bw, const float* bias,
                         float* Cf, __bf16* Cb, int flags, int dy, int dx) {
        dim3 grid((unsigned)(256 / 64), (unsigned)(MT / 64));
        dim3 block(32, 4);
        gemm_bf16_kernel<true><<<grid, block, 0, stream>>>(A, 256, Bw, 256, bias,
                                                           Cf, 256, Cb, 256,
                                                           (int)MT, 256, 256, flags, dy, dx);
    };
    auto ln = [&](const float* Y, const float* R, const float* g, const float* b,
                  float* oF, __bf16* o0, long ld0, __bf16* o1, long ld1, int M) {
        dim3 block(32, 8);
        ln256_kernel<<<(M + 7) / 8, block, 0, stream>>>(Y, R, g, b, oF, o0, ld0, o1, ld1, M);
    };

    // ---- 1) weight conversion / repack + zero recurrent state ----
    cvt(in_proj_w, inpw, 768 * 256);
    cvt(out_proj_w, outpw, 256 * 256);
    cvt(ff1_w, ff1w, 512 * 256);
    cvt(ff2_w, ff2w, 256 * 512);
    cvt(be1_w, be1w, (long)256 * 16384);
    cvt(be2_w, be2w, 256 * 256);
    cvt(L1_w, L1wB, 256 * 768);
    cvt(L2_w, L2wB, 256 * 256);
    cvt(H1_w, H1wB, 256 * 512);
    cvt(H2_w, H2wB, 256 * 256);
    cvt(v1_w, v1wB, 256 * 256);
    cvt(v2_w, v2wB, 128 * 256);
    {
        long n = (long)256 * 256 * 9;
        repack_conv2_kernel<<<(unsigned)((n + 255) / 256), 256, 0, stream>>>(conv2_w, w2p);
    }
    hipMemsetAsync(catL, 0, (size_t)BATCH * 768 * 2, stream);  // zL=zH=0, be filled later
    hipMemsetAsync(catH, 0, (size_t)BATCH * 512 * 2, stream);

    // ---- 2) conv1 (direct VALU) ----
    {
        long n = MT * 256;
        conv1_kernel<<<(unsigned)((n + 255) / 256), 256, 0, stream>>>(bitplanes, conv1_w, conv1_b, c1b);
    }

    // ---- 3) conv2 as 9 gather-GEMM passes over (dy,dx) ----
    {
        int pass = 0;
        for (int dy = -1; dy <= 1; dy++) {
            for (int dx = -1; dx <= 1; dx++) {
                int flags = FLAG_WF32;
                const float* bias = nullptr;
                if (pass > 0) flags |= FLAG_ACC;
                if (pass == 8) { flags |= FLAG_RELU | FLAG_WBF16; bias = conv2_b; }
                gemm_conv(c1b, w2p + (long)pass * 65536, bias, x0f, x0b, flags, dy, dx);
                pass++;
            }
        }
    }

    // ---- 4) transformer layer ----
    // qkv = x @ in_proj^T + b
    gemm(x0b, 256, inpw, 256, in_proj_b, qkvf, 768, nullptr, 0,
         (int)MT, 768, 256, FLAG_WF32);
    // fused attention (8192 WGs: one per (b,h))
    attn_kernel<<<8192, 256, 0, stream>>>(qkvf, obb);
    // out_proj
    gemm(obb, 256, outpw, 256, out_proj_b, tmpf, 256, nullptr, 0,
         (int)MT, 256, 256, FLAG_WF32);
    // LN1(x + o)
    ln(tmpf, x0f, ln1_g, ln1_b, x1f, x1b, 256, nullptr, 0, (int)MT);
    // FF
    gemm(x1b, 256, ff1w, 256, ff1_b, nullptr, 0, ffh, 512,
         (int)MT, 512, 256, FLAG_RELU | FLAG_WBF16);
    gemm(ffh, 512, ff2w, 512, ff2_b, tmpf, 256, nullptr, 0,
         (int)MT, 256, 512, FLAG_WF32);
    // LN2(x + ff) -> x2 (bf16 only; reinterpreted as xf [2048,16384])
    ln(tmpf, x1f, ln2_g, ln2_b, nullptr, x2b, 256, nullptr, 0, (int)MT);

    // ---- 5) board embedding: be = relu(relu(xf@be1^T+b)@be2^T+b) -> catL[:,512:768]
    gemm(x2b, 16384, be1w, 16384, be1_b, nullptr, 0, beh, 256,
         (int)BATCH, 256, 16384, FLAG_RELU | FLAG_WBF16);
    gemm(beh, 256, be2w, 256, be2_b, nullptr, 0, catL + 512, 768,
         (int)BATCH, 256, 256, FLAG_RELU | FLAG_WBF16);

    // ---- 6) recurrent HRM loop ----
    auto Lnet = [&]() {
        gemm(catL, 768, L1wB, 768, L1_b, nullptr, 0, hidb, 256,
             (int)BATCH, 256, 768, FLAG_RELU | FLAG_WBF16);
        gemm(hidb, 256, L2wB, 256, L2_b, tmp2f, 256, nullptr, 0,
             (int)BATCH, 256, 256, FLAG_RELU | FLAG_WF32);
        // zL -> catL[:,0:256] and catH[:,256:512]
        ln(tmp2f, nullptr, Lln_g, Lln_b, nullptr, catL, 768, catH + 256, 512, (int)BATCH);
    };
    auto Hnet = [&]() {
        gemm(catH, 512, H1wB, 512, H1_b, nullptr, 0, hidb, 256,
             (int)BATCH, 256, 512, FLAG_RELU | FLAG_WBF16);
        gemm(hidb, 256, H2wB, 256, H2_b, tmp2f, 256, nullptr, 0,
             (int)BATCH, 256, 256, FLAG_RELU | FLAG_WF32);
        // zH -> catL[:,256:512] and catH[:,0:256]
        ln(tmp2f, nullptr, Hln_g, Hln_b, nullptr, catL + 256, 768, catH, 512, (int)BATCH);
    };

    for (int i = 0; i < 63; i++) {
        Lnet();
        if (((i + 1) & 7) == 0) Hnet();  // H update every T=8 steps
    }
    Lnet();
    Hnet();

    // ---- 7) value head: relu(zH@v1^T+b)@v2^T+b -> d_out [2048,128] ----
    gemm(catH, 512, v1wB, 256, v1_b, nullptr, 0, hid2b, 256,
         (int)BATCH, 256, 256, FLAG_RELU | FLAG_WBF16);      // zH = catH[:,0:256]
    gemm(hid2b, 256, v2wB, 256, v2_b, outF, 128, nullptr, 0,
         (int)BATCH, 128, 256, FLAG_WF32);
}
